// PhaseAttention_44693429682850
// MI455X (gfx1250) — matrix operand order repaired and checked by reading, not compiled, not run on hardware
//
#include <hip/hip_runtime.h>
#include <hip/hip_bf16.h>

// ---------------------------------------------------------------------------
// PhaseAttention for MI455X (gfx1250): bf16 WMMA path (v_wmma_f32_16x16x32_bf16)
// ---------------------------------------------------------------------------

typedef __attribute__((ext_vector_type(16))) __bf16 bf16x16;
typedef __attribute__((ext_vector_type(8)))  __bf16 bf16x8;
typedef __attribute__((ext_vector_type(8)))  float  f32x8;

#define B_    2
#define L_    2048
#define DIM_  1024
#define H_    16
#define HD_   64
#define WIN_  256
#define M_    (B_ * L_)     // 4096 rows for the big GEMMs
#define K2_   (2 * DIM_)    // 2048 (real|imag stacked K)
#define N2_   (2 * DIM_)    // 2048 (real|imag stacked N)
#define SCALE_ 0.125f       // HD^-0.5
#define NEGV  (-1.0e30f)

// ---------------------------------------------------------------------------
// Fragment loaders, following CDNA5 ISA 7.12.2 VGPR layouts (wave32).
// A 16x32 bf16: lanes 0-15 -> row=lane, K {0..7,16..23}; lanes 16-31 -> row=lane-16,
//               K {8..15,24..31}. Two 16B chunks per lane.
// B 32x16 bf16 (stored N-major, stride elements per column): lanes 0-15 -> col=lane,
//               K {0..7,16..23}; lanes 16-31 -> col=lane-16, K {8..15,24..31}. Two 16B chunks per lane.
// ---------------------------------------------------------------------------
__device__ __forceinline__ bf16x16 frag_a_ld(const __bf16* p, int stride, int lane) {
    const int r  = lane & 15;
    const int kh = (lane >> 4) << 3;       // 0 or 8
    union { bf16x16 v; bf16x8 h[2]; } u;
    u.h[0] = *(const bf16x8*)(p + (size_t)r * stride + kh);
    u.h[1] = *(const bf16x8*)(p + (size_t)r * stride + kh + 16);
    return u.v;
}

__device__ __forceinline__ bf16x16 frag_b_nmajor(const __bf16* p, int stride, int lane) {
    const int c  = lane & 15;
    const int kh = (lane >> 4) << 3;       // 0 or 8
    union { bf16x16 v; bf16x8 h[2]; } u;
    u.h[0] = *(const bf16x8*)(p + (size_t)c * stride + kh);
    u.h[1] = *(const bf16x8*)(p + (size_t)c * stride + kh + 16);
    return u.v;
}

__device__ __forceinline__ f32x8 wmma_bf16(bf16x16 a, bf16x16 b, f32x8 c) {
    return __builtin_amdgcn_wmma_f32_16x16x32_bf16(false, a, false, b,
                                                   (short)0, c, false, false);
}

// ---------------------------------------------------------------------------
// Prep kernels: fp32 -> bf16 repack
// ---------------------------------------------------------------------------
// Az[m][k] for m in [0,4096), k in [0,2048): k<1024 -> Re z, else Im z
__global__ void prep_z_kernel(const float* __restrict__ z, __bf16* __restrict__ Az) {
    size_t idx = (size_t)blockIdx.x * 256 + threadIdx.x;     // 4096*2048 total
    int m = (int)(idx >> 11);
    int k = (int)(idx & 2047);
    int d = k & 1023, c = k >> 10;
    Az[idx] = (__bf16)z[((size_t)m * DIM_ + d) * 2 + c];
}

// Bm = [[Wr, Wi], [-Wi, Wr]]  (2048 x 2048), row = input dim, col = output dim
__global__ void prep_w_kernel(const float* __restrict__ W, __bf16* __restrict__ Bm) {
    size_t idx = (size_t)blockIdx.x * 256 + threadIdx.x;     // 2048*2048 total
    int k = (int)(idx >> 11);
    int n = (int)(idx & 2047);
    int c = ((k < DIM_) != (n < DIM_)) ? 1 : 0;
    float s = (k >= DIM_ && n < DIM_) ? -1.0f : 1.0f;
    Bm[idx] = (__bf16)(s * W[((size_t)(k & 1023) * DIM_ + (n & 1023)) * 2 + c]);
}

// ---------------------------------------------------------------------------
// Complex GEMM: C[4096 x 2048] = A[4096 x 2048] * Bm[2048 x 2048] (bf16, f32 acc)
// 128x128 tile per 256-thread block; 8 waves in 2(m) x 4(n); 8 WMMA tiles/wave/kstep.
// MODE 0: scatter to q/k planes [b,h,l,hd] (real plane / imag plane)
// MODE 1: scatter to v planes TRANSPOSED [b,h,hd,l]
// MODE 2: fp32 complex-interleaved output [m][d][2]
// ---------------------------------------------------------------------------
template<int MODE>
__global__ __launch_bounds__(256)
void gemm_cplx_kernel(const __bf16* __restrict__ A, const __bf16* __restrict__ Bm,
                      __bf16* __restrict__ outR, __bf16* __restrict__ outI,
                      float* __restrict__ outF)
{
    __shared__ __align__(16) __bf16 As[128 * 40];   // row-major, stride 40 (16B aligned)
    __shared__ __align__(16) __bf16 Bs[128 * 40];   // n-major:  Bs[n*40 + k]

    const int tid  = threadIdx.x;
    const int lane = tid & 31;
    const int wave = tid >> 5;
    const int wm   = wave & 1;        // 2 m-groups of 64
    const int wn   = wave >> 1;       // 4 n-groups of 32
    const int mBase = blockIdx.y * 128;
    const int nBase = blockIdx.x * 128;

    f32x8 acc[4][2];
#pragma unroll
    for (int i = 0; i < 4; ++i)
#pragma unroll
        for (int j = 0; j < 2; ++j)
#pragma unroll
            for (int e = 0; e < 8; ++e) acc[i][j][e] = 0.0f;

    for (int k0 = 0; k0 < K2_; k0 += 32) {
        // ---- stage A tile 128x32 (16B chunks) ----
#pragma unroll
        for (int t = 0; t < 2; ++t) {
            int c   = tid + t * 256;            // 0..511 chunks
            int row = c >> 2;
            int kc  = (c & 3) << 3;
            *(bf16x8*)(&As[row * 40 + kc]) =
                *(const bf16x8*)(A + (size_t)(mBase + row) * K2_ + k0 + kc);
        }
        // ---- stage B tile 32x128, transposed to n-major ----
#pragma unroll
        for (int t = 0; t < 2; ++t) {
            int c  = tid + t * 256;
            int kr = c >> 4;                    // 0..31
            int nc = (c & 15) << 3;             // 0..120
            union { bf16x8 v; __bf16 e[8]; } u;
            u.v = *(const bf16x8*)(Bm + (size_t)(k0 + kr) * N2_ + nBase + nc);
#pragma unroll
            for (int j = 0; j < 8; ++j) Bs[(nc + j) * 40 + kr] = u.e[j];
        }
        __syncthreads();

        bf16x16 af[4], bfm[2];
#pragma unroll
        for (int mt = 0; mt < 4; ++mt)
            af[mt] = frag_a_ld(&As[(wm * 64 + mt * 16) * 40], 40, lane);
#pragma unroll
        for (int nt = 0; nt < 2; ++nt)
            bfm[nt] = frag_b_nmajor(&Bs[(wn * 32 + nt * 16) * 40], 40, lane);
#pragma unroll
        for (int mt = 0; mt < 4; ++mt)
#pragma unroll
            for (int nt = 0; nt < 2; ++nt)
                acc[mt][nt] = wmma_bf16(af[mt], bfm[nt], acc[mt][nt]);
        __syncthreads();
    }

    // ---- epilogue (C layout: lanes 0-15 VGPR j -> row j; lanes 16-31 -> row j+8) ----
    const int half  = lane >> 4;
    const int col15 = lane & 15;
#pragma unroll
    for (int mt = 0; mt < 4; ++mt) {
#pragma unroll
        for (int nt = 0; nt < 2; ++nt) {
#pragma unroll
            for (int j = 0; j < 8; ++j) {
                int m = mBase + wm * 64 + mt * 16 + j + half * 8;
                int n = nBase + wn * 32 + nt * 16 + col15;
                float v = acc[mt][nt][j];
                int d = n & 1023, cc = n >> 10;
                if (MODE == 2) {
                    outF[((size_t)m * DIM_ + d) * 2 + cc] = v;
                } else {
                    int h = d >> 6, hd = d & 63;
                    int b = m >> 11, lq = m & 2047;
                    size_t bh = (size_t)(b * H_ + h);
                    __bf16  bv  = (__bf16)v;
                    __bf16* dst = cc ? outI : outR;
                    if (MODE == 0) dst[(bh * L_ + lq) * HD_ + hd] = bv; // [b,h,l,hd]
                    else           dst[(bh * HD_ + hd) * L_ + lq] = bv; // [b,h,hd,l]
                }
            }
        }
    }
}

// ---------------------------------------------------------------------------
// Sliding-window flash attention. Block = 256 thr = 8 waves; each wave owns 16
// queries of one (b,h). Q frags / K^T frags / V^T frags load straight from
// global (layouts match WMMA VGPR striping); only P goes through LDS (C->A
// layout conversion), ordered with an explicit s_wait_dscnt.
// ---------------------------------------------------------------------------
__global__ __launch_bounds__(256)
void attn_window_kernel(const __bf16* __restrict__ qr, const __bf16* __restrict__ qi,
                        const __bf16* __restrict__ kr, const __bf16* __restrict__ ki,
                        const __bf16* __restrict__ vrT, const __bf16* __restrict__ viT,
                        __bf16* __restrict__ attnA)
{
    __shared__ __align__(16) __bf16 Pbuf[8][16 * 40];

    const int tid   = threadIdx.x;
    const int lane  = tid & 31;
    const int wave  = tid >> 5;
    const int half  = lane >> 4;
    const int col15 = lane & 15;
    const int bh    = blockIdx.y;                 // 0..31  (b*16 + h)
    const int q0    = blockIdx.x * 128 + wave * 16;
    const size_t plane = (size_t)bh * L_ * HD_;   // per-plane (b,h) offset

    // Q fragments: 16 queries x HD, split into two K=32 halves, per component
    bf16x16 fqr[2], fqi[2];
#pragma unroll
    for (int s = 0; s < 2; ++s) {
        fqr[s] = frag_a_ld(qr + plane + (size_t)q0 * HD_ + s * 32, HD_, lane);
        fqi[s] = frag_a_ld(qi + plane + (size_t)q0 * HD_ + s * 32, HD_, lane);
    }

    f32x8 O[2][4];
    float mrow[8], lrow[8];
#pragma unroll
    for (int j = 0; j < 8; ++j) { mrow[j] = NEGV; lrow[j] = 0.0f; }
#pragma unroll
    for (int c = 0; c < 2; ++c)
#pragma unroll
        for (int t = 0; t < 4; ++t)
#pragma unroll
            for (int e = 0; e < 8; ++e) O[c][t][e] = 0.0f;

    const int qEnd = q0 + 15;
    int kb0 = q0 - (WIN_ - 1);
    if (kb0 < 0) kb0 = 0;
    kb0 &= ~31;                                   // 32-key blocks; mask handles slack

    for (int kb = kb0; kb <= qEnd; kb += 32) {
        // ---- scores: two 16-key sub-tiles, 4 WMMAs each (qr*kr + qi*ki) ----
        f32x8 S[2];
#pragma unroll
        for (int t = 0; t < 2; ++t) {
            int kbb = kb + t * 16;
            f32x8 s;
#pragma unroll
            for (int e = 0; e < 8; ++e) s[e] = 0.0f;
#pragma unroll
            for (int hs = 0; hs < 2; ++hs) {
                bf16x16 fkr = frag_b_nmajor(kr + plane + (size_t)kbb * HD_ + hs * 32, HD_, lane);
                s = wmma_bf16(fqr[hs], fkr, s);
                bf16x16 fki = frag_b_nmajor(ki + plane + (size_t)kbb * HD_ + hs * 32, HD_, lane);
                s = wmma_bf16(fqi[hs], fki, s);
            }
            int key = kbb + col15;                // this lane's key column
#pragma unroll
            for (int j = 0; j < 8; ++j) {
                int qq  = q0 + j + half * 8;      // this element's query row
                bool ok = (key <= qq) && ((qq - key) < WIN_);
                s[j] = ok ? (s[j] * SCALE_) : NEGV;
            }
            S[t] = s;
        }

        // ---- online softmax: row max / rescale / exp / row sum ----
        float mnew[8], alpha[8];
#pragma unroll
        for (int j = 0; j < 8; ++j) {
            float mx = fmaxf(S[0][j], S[1][j]);
#pragma unroll
            for (int d = 1; d < 16; d <<= 1) mx = fmaxf(mx, __shfl_xor(mx, d, 32));
            mnew[j]  = fmaxf(mrow[j], mx);
            alpha[j] = __expf(mrow[j] - mnew[j]);
            mrow[j]  = mnew[j];
        }
#pragma unroll
        for (int t = 0; t < 2; ++t) {
#pragma unroll
            for (int j = 0; j < 8; ++j) {
                float p = __expf(S[t][j] - mnew[j]);
                S[t][j] = p;
                Pbuf[wave][(j + half * 8) * 40 + t * 16 + col15] = (__bf16)p;
            }
        }
#pragma unroll
        for (int j = 0; j < 8; ++j) {
            float rs = S[0][j] + S[1][j];
#pragma unroll
            for (int d = 1; d < 16; d <<= 1) rs += __shfl_xor(rs, d, 32);
            lrow[j] = lrow[j] * alpha[j] + rs;
        }
#pragma unroll
        for (int c = 0; c < 2; ++c)
#pragma unroll
            for (int t = 0; t < 4; ++t)
#pragma unroll
                for (int j = 0; j < 8; ++j) O[c][t][j] *= alpha[j];

        // order per-wave LDS stores (C-layout P) before A-layout reload
        asm volatile("s_wait_dscnt 0" ::: "memory");
        bf16x16 fp = frag_a_ld(&Pbuf[wave][0], 40, lane);

        // ---- O += P(16x32) @ V(32x64), per component: 8 WMMAs ----
#pragma unroll
        for (int t = 0; t < 4; ++t) {
            bf16x16 fvr = frag_b_nmajor(vrT + plane + (size_t)(t * 16) * L_ + kb, L_, lane);
            O[0][t] = wmma_bf16(fp, fvr, O[0][t]);
            bf16x16 fvi = frag_b_nmajor(viT + plane + (size_t)(t * 16) * L_ + kb, L_, lane);
            O[1][t] = wmma_bf16(fp, fvi, O[1][t]);
        }
    }

    // ---- normalize and write attention output as A matrix for Wo GEMM ----
    const int b = bh >> 4, h = bh & 15;
#pragma unroll
    for (int c = 0; c < 2; ++c)
#pragma unroll
        for (int t = 0; t < 4; ++t)
#pragma unroll
            for (int j = 0; j < 8; ++j) {
                float v = O[c][t][j] / lrow[j];
                int qq  = q0 + j + half * 8;
                int m   = b * L_ + qq;
                int col = c * DIM_ + h * HD_ + t * 16 + col15;
                attnA[(size_t)m * K2_ + col] = (__bf16)v;
            }
}

// ---------------------------------------------------------------------------
// Host launcher. Inputs: z, Wq, Wk, Wv, Wo (fp32). Output: [B,L,DIM,2] fp32.
// Workspace layout (MiB offsets), total 96 MiB:
//   0: Az (16)  16: Bq (8)  24: Bk (8)  32: Bv (8)  40: Bo (8)
//  48: qr  56: qi  64: kr  72: ki  80: vrT  88: viT   (8 each)
//  attnA reuses Az (z staging dead after the QKV GEMMs).
// ---------------------------------------------------------------------------
extern "C" void kernel_launch(void* const* d_in, const int* in_sizes, int n_in,
                              void* d_out, int out_size, void* d_ws, size_t ws_size,
                              hipStream_t stream) {
    (void)in_sizes; (void)n_in; (void)out_size; (void)ws_size;
    const float* z  = (const float*)d_in[0];
    const float* Wq = (const float*)d_in[1];
    const float* Wk = (const float*)d_in[2];
    const float* Wv = (const float*)d_in[3];
    const float* Wo = (const float*)d_in[4];
    float* out = (float*)d_out;

    char* ws = (char*)d_ws;
    const size_t MB = (size_t)1 << 20;
    __bf16* Az  = (__bf16*)(ws + 0 * MB);
    __bf16* Bq  = (__bf16*)(ws + 16 * MB);
    __bf16* Bk  = (__bf16*)(ws + 24 * MB);
    __bf16* Bv  = (__bf16*)(ws + 32 * MB);
    __bf16* Bo  = (__bf16*)(ws + 40 * MB);
    __bf16* qr  = (__bf16*)(ws + 48 * MB);
    __bf16* qi  = (__bf16*)(ws + 56 * MB);
    __bf16* kr  = (__bf16*)(ws + 64 * MB);
    __bf16* ki  = (__bf16*)(ws + 72 * MB);
    __bf16* vrT = (__bf16*)(ws + 80 * MB);
    __bf16* viT = (__bf16*)(ws + 88 * MB);
    __bf16* attnA = Az;                    // reuse: Az dead after QKV GEMMs

    // 1) repack inputs to bf16
    prep_z_kernel<<<(M_ * K2_) / 256, 256, 0, stream>>>(z, Az);
    prep_w_kernel<<<(K2_ * N2_) / 256, 256, 0, stream>>>(Wq, Bq);
    prep_w_kernel<<<(K2_ * N2_) / 256, 256, 0, stream>>>(Wk, Bk);
    prep_w_kernel<<<(K2_ * N2_) / 256, 256, 0, stream>>>(Wv, Bv);
    prep_w_kernel<<<(K2_ * N2_) / 256, 256, 0, stream>>>(Wo, Bo);

    // 2) complex projections (WMMA GEMMs)
    dim3 ggrid(N2_ / 128, M_ / 128);       // (16, 32)
    gemm_cplx_kernel<0><<<ggrid, 256, 0, stream>>>(Az, Bq, qr, qi, nullptr);
    gemm_cplx_kernel<0><<<ggrid, 256, 0, stream>>>(Az, Bk, kr, ki, nullptr);
    gemm_cplx_kernel<1><<<ggrid, 256, 0, stream>>>(Az, Bv, vrT, viT, nullptr);

    // 3) sliding-window attention (WMMA flash)
    dim3 agrid(L_ / 128, B_ * H_);         // (16, 32)
    attn_window_kernel<<<agrid, 256, 0, stream>>>(qr, qi, kr, ki, vrT, viT, attnA);

    // 4) output projection, fp32 complex-interleaved result
    gemm_cplx_kernel<2><<<ggrid, 256, 0, stream>>>(attnA, Bo, nullptr, nullptr, out);
}
